// MixtureOfExperts_67010079752265
// MI455X (gfx1250) — compile-verified, matrix operand order checked by reading, not run on hardware
//
#include <hip/hip_runtime.h>
#include <hip/hip_bf16.h>

#define NTOK 8192
#define DDIM 1024
#define HDIM 4096
#define NEXP 4

typedef __attribute__((ext_vector_type(16))) __bf16 v16bf;
typedef __attribute__((ext_vector_type(8)))  __bf16 v8bf;
typedef __attribute__((ext_vector_type(8)))  float  v8f;
typedef __attribute__((ext_vector_type(4)))  float  v4f;

// LDS row strides (bf16 units): rows 16B-aligned, dword-stride % 64 == 4 spreads banks
#define XSS 1032   // x tile : 32 x 1032 (66,048 B)
#define HSS 136    // h tile : 32 x 136  ( 8,704 B)

// workspace layout (bytes)
#define OFF_CNT  ((size_t)NTOK * NEXP * 4)                    // 131072
#define OFF_LIST (OFF_CNT + 128)                              // 131200
#define OFF_W1T  ((size_t)4096 * 65)                          // 266240 (4KB aligned)
#define OFF_W2T  (OFF_W1T + (size_t)NEXP * DDIM * HDIM * 2)   // + 32 MB

static __device__ __forceinline__ unsigned pack2(float a, float b) {
  union { __bf16 h[2]; unsigned u; } u;
  u.h[0] = (__bf16)a; u.h[1] = (__bf16)b;
  return u.u;
}

static __device__ __forceinline__ v16bf ld_frag(const __bf16* row, int o0, int o1) {
  v8bf lo = *(const v8bf*)(row + o0);   // ds_load_b128 / global_load_b128
  v8bf hi = *(const v8bf*)(row + o1);
  return __builtin_shufflevector(lo, hi, 0,1,2,3,4,5,6,7,8,9,10,11,12,13,14,15);
}

#define WMMA_BF16(a, b, c) \
  __builtin_amdgcn_wmma_f32_16x16x32_bf16(false, (a), false, (b), (short)0, (c), false, false)

static __device__ __forceinline__ void atomAddF(float* p, float v) {
  __hip_atomic_fetch_add(p, v, __ATOMIC_RELAXED, __HIP_MEMORY_SCOPE_AGENT);
}

// ------- One-time per launch: f32 [E][R][C] -> bf16 transposed [E][C][R] ----
__global__ __launch_bounds__(256) void transpose_bf16(const float* __restrict__ src,
                                                      __bf16* __restrict__ dst,
                                                      int R, int C) {
  __shared__ float t[32][33];
  const size_t so = (size_t)blockIdx.z * R * C;
  const int c0 = blockIdx.x * 32, r0 = blockIdx.y * 32;
  const int tx = threadIdx.x & 31, ty = threadIdx.x >> 5;
#pragma unroll
  for (int i = 0; i < 4; ++i)
    t[ty + 8 * i][tx] = src[so + (size_t)(r0 + ty + 8 * i) * C + c0 + tx];
  __syncthreads();
  const int cc = threadIdx.x >> 4;        // 0..15
  const int rp = (threadIdx.x & 15) * 2;  // even r pairs
#pragma unroll
  for (int i = 0; i < 2; ++i) {
    int c = cc + 16 * i;
    unsigned v = pack2(t[rp][c], t[rp + 1][c]);
    *(unsigned*)(dst + so + (size_t)(c0 + c) * R + r0 + rp) = v;
  }
}

// ---------------- Router: one wave (32 lanes) per token -------------------
__global__ __launch_bounds__(256) void moe_router(const float* __restrict__ x,
                                                  const float* __restrict__ Wr,
                                                  float* __restrict__ wts) {
  const int tok  = (blockIdx.x * 256 + threadIdx.x) >> 5;
  const int lane = threadIdx.x & 31;
  if (tok >= NTOK) return;
  const float* xr = x + (size_t)tok * DDIM;
  float a0 = 0.f, a1 = 0.f, a2 = 0.f, a3 = 0.f;
  for (int i = 0; i < 8; ++i) {
    int d = (i * 32 + lane) * 4;
    v4f xv = *(const v4f*)(xr + d);
#pragma unroll
    for (int j = 0; j < 4; ++j) {
      v4f wv = *(const v4f*)(Wr + (size_t)(d + j) * NEXP);
      a0 += xv[j] * wv[0];
      a1 += xv[j] * wv[1];
      a2 += xv[j] * wv[2];
      a3 += xv[j] * wv[3];
    }
  }
#pragma unroll
  for (int o = 16; o > 0; o >>= 1) {
    a0 += __shfl_xor(a0, o, 32);
    a1 += __shfl_xor(a1, o, 32);
    a2 += __shfl_xor(a2, o, 32);
    a3 += __shfl_xor(a3, o, 32);
  }
  float l[NEXP] = {a0, a1, a2, a3};
  float m = fmaxf(fmaxf(l[0], l[1]), fmaxf(l[2], l[3]));
  float p[NEXP];
#pragma unroll
  for (int e = 0; e < NEXP; ++e) p[e] = __expf(l[e] - m);
  int i1 = 0;
#pragma unroll
  for (int e = 1; e < NEXP; ++e) if (p[e] > p[i1]) i1 = e;
  int i2 = (i1 == 0) ? 1 : 0;
#pragma unroll
  for (int e = 0; e < NEXP; ++e) if (e != i1 && p[e] > p[i2]) i2 = e;
  float s = p[i1] + p[i2];
  if (lane < NEXP) {
    float w = (lane == i1 || lane == i2) ? p[lane] / s : 0.0f;
    wts[tok * NEXP + lane] = w;
  }
}

// -- Order-preserving token->expert compaction: one block per expert -------
__global__ __launch_bounds__(256) void moe_compact(const float* __restrict__ wts,
                                                   int* __restrict__ cnt,
                                                   int* __restrict__ list) {
  __shared__ int woff[8];
  __shared__ int sbase;
  const int e = blockIdx.x;
  const int tid = threadIdx.x, wv = tid >> 5, lane = tid & 31;
  if (tid == 0) sbase = 0;
  __syncthreads();
  for (int c = 0; c < NTOK; c += 256) {
    int tok = c + tid;
    bool act = wts[tok * NEXP + e] != 0.0f;
    unsigned long long bal = __ballot(act);
    if (lane == 0) woff[wv] = __popcll(bal);
    __syncthreads();
    if (tid == 0) {                       // tiny serial scan of 8 wave sums
      int s = sbase;
#pragma unroll
      for (int i = 0; i < 8; ++i) { int v = woff[i]; woff[i] = s; s += v; }
      sbase = s;
    }
    __syncthreads();
    if (act) {
      int pos = woff[wv] + __popcll(bal & ((1ull << lane) - 1ull));
      list[e * NTOK + pos] = tok;
    }
    __syncthreads();                      // woff reused next chunk
  }
  if (tid == 0) cnt[e] = sbase;
}

// ---- Fused expert FFN: M=32 gathered tokens, bf16 WMMA, f32 accumulate ---
// H processed in 128-wide chunks; each wave owns one exclusive 16-col N-tile
// of the h tile (computes both 16-row M-tiles), so every weight fragment is
// loaded exactly once per block (global_load_b128, K-contiguous layout).
__global__ __launch_bounds__(256) void moe_ffn(const float* __restrict__ x,
                                               const __bf16* __restrict__ W1t,  // [E][H][D]
                                               const float* __restrict__ b1,
                                               const __bf16* __restrict__ W2t,  // [E][D][H]
                                               const float* __restrict__ b2,
                                               const float* __restrict__ wts,
                                               const int* __restrict__ cnt,
                                               const int* __restrict__ list,
                                               float* __restrict__ out) {
  __shared__ __bf16 xs[32 * XSS];
  __shared__ __bf16 hs[32 * HSS];
  __shared__ float b1s[128];
  __shared__ float wLds[32];
  __shared__ int   tLds[32];

  const int e = blockIdx.y;
  const int nTok = cnt[e];
  const int tb = blockIdx.x * 32;
  if (tb >= nTok) return;           // uniform across block

  const int tid = threadIdx.x;
  const int waveId = tid >> 5;
  const int lane = tid & 31;
  const int laneLo = lane & 15;
  const int hi8 = (lane >> 4) * 8;

  if (tid < 32) {
    int idx = tb + tid;
    int tok = (idx < nTok) ? list[e * NTOK + idx] : 0;
    tLds[tid] = tok;
    wLds[tid] = (idx < nTok) ? wts[tok * NEXP + e] : 0.0f;  // pad rows weight=0
  }
  __syncthreads();

  // gather x tile -> bf16 LDS (32 x 1024); iteration p: whole block on row p
  for (int p = 0; p < 32; ++p) {
    int c4 = tid;                          // float4 column
    v4f xv = *(const v4f*)(x + (size_t)tLds[p] * DDIM + c4 * 4);
    unsigned* dst = (unsigned*)(xs + p * XSS + c4 * 4);
    dst[0] = pack2(xv[0], xv[1]);
    dst[1] = pack2(xv[2], xv[3]);
  }
  __syncthreads();

  v8f z = {};
  v8f acc[4][2][2];               // [dchunk][mtile][ntile], 128 VGPRs/lane
#pragma unroll
  for (int i = 0; i < 4; ++i)
#pragma unroll
    for (int j = 0; j < 2; ++j)
#pragma unroll
      for (int k = 0; k < 2; ++k) acc[i][j][k] = z;

  const __bf16* pa0 = xs + laneLo * XSS;          // A rows, M-tile 0 (LDS)
  const __bf16* pa1 = xs + (16 + laneLo) * XSS;   // A rows, M-tile 1 (LDS)
  // GEMM1 B rows: wave owns exclusive N-tile = waveId (16 h-columns)
  const __bf16* pb1 = W1t + ((size_t)e * HDIM + waveId * 16 + laneLo) * DDIM;

  for (int hc = 0; hc < HDIM / 128; ++hc) {
    const int h0 = hc * 128;
    if (tid < 128) b1s[tid] = b1[(size_t)e * HDIM + h0 + tid];  // prefetch bias

    // warm next H-chunk's B row (stream discontinuity); in-bounds only
    if (hc + 1 < HDIM / 128)
      __builtin_prefetch(pb1 + (size_t)(h0 + 128) * DDIM, 0, 3);

    // ----- GEMM1: h(32x128) = x @ W1[:, h0:h0+128], K=1024, 2 acc chains ---
    v8f c1a = z, c1b = z;
    const __bf16* pbr = pb1 + (size_t)h0 * DDIM;
    for (int kb = 0; kb < DDIM; kb += 128) {
#pragma unroll
      for (int k2 = 0; k2 < 4; ++k2) {
        int kk = kb + k2 * 32;
        v16bf b  = ld_frag(pbr, kk + hi8, kk + 16 + hi8);
        v16bf a0 = ld_frag(pa0, kk + hi8, kk + 16 + hi8);
        v16bf a1 = ld_frag(pa1, kk + hi8, kk + 16 + hi8);
        c1a = WMMA_BF16(a0, b, c1a);
        c1b = WMMA_BF16(a1, b, c1b);
      }
    }
    __syncthreads();   // b1s ready; prior hc's hs readers are done

    // bias + ReLU -> bf16 h tile in LDS (wave writes its exclusive 16 cols)
    {
      int ccol = waveId * 16 + laneLo;
      float b1v = b1s[ccol];
#pragma unroll
      for (int v = 0; v < 8; ++v) {
        float ha = c1a[v] + b1v;
        float hb = c1b[v] + b1v;
        ha = ha > 0.0f ? ha : 0.0f;
        hb = hb > 0.0f ? hb : 0.0f;
        hs[(hi8 + v) * HSS + ccol]      = (__bf16)ha;
        hs[(16 + hi8 + v) * HSS + ccol] = (__bf16)hb;
      }
    }
    __syncthreads();   // hs ready

    // ----- GEMM2: y += h @ W2[h0:h0+128, :], N=1024 ------------------------
#pragma unroll
    for (int ks = 0; ks < 4; ++ks) {
      v16bf a0 = ld_frag(hs + laneLo * HSS,        ks * 32 + hi8, ks * 32 + 16 + hi8);
      v16bf a1 = ld_frag(hs + (16 + laneLo) * HSS, ks * 32 + hi8, ks * 32 + 16 + hi8);
      int ko = h0 + ks * 32;
#pragma unroll
      for (int dc = 0; dc < 4; ++dc) {
        const __bf16* pb2 =
            W2t + ((size_t)e * DDIM + dc * 256 + waveId * 32 + laneLo) * HDIM;
        v16bf b0  = ld_frag(pb2,             ko + hi8, ko + 16 + hi8);
        v16bf b1f = ld_frag(pb2 + 16 * HDIM, ko + hi8, ko + 16 + hi8);
        acc[dc][0][0] = WMMA_BF16(a0, b0,  acc[dc][0][0]);
        acc[dc][0][1] = WMMA_BF16(a0, b1f, acc[dc][0][1]);
        acc[dc][1][0] = WMMA_BF16(a1, b0,  acc[dc][1][0]);
        acc[dc][1][1] = WMMA_BF16(a1, b1f, acc[dc][1][1]);
      }
    }
  }

  // epilogue: out[t, n] += w_t * (acc + b2[n])  (exactly 2 adds per element)
#pragma unroll
  for (int dc = 0; dc < 4; ++dc)
#pragma unroll
    for (int mt = 0; mt < 2; ++mt)
#pragma unroll
      for (int nt = 0; nt < 2; ++nt) {
        int n = dc * 256 + waveId * 32 + nt * 16 + laneLo;
        float b2v = b2[(size_t)e * DDIM + n];
#pragma unroll
        for (int v = 0; v < 8; ++v) {
          int rl = mt * 16 + hi8 + v;
          float val = (acc[dc][mt][nt][v] + b2v) * wLds[rl];
          atomAddF(out + (size_t)tLds[rl] * DDIM + n, val);
        }
      }
}

extern "C" void kernel_launch(void* const* d_in, const int* in_sizes, int n_in,
                              void* d_out, int out_size, void* d_ws, size_t ws_size,
                              hipStream_t stream) {
  const float* x  = (const float*)d_in[0];
  const float* Wr = (const float*)d_in[1];
  const float* W1 = (const float*)d_in[2];
  const float* b1 = (const float*)d_in[3];
  const float* W2 = (const float*)d_in[4];
  const float* b2 = (const float*)d_in[5];
  float* out = (float*)d_out;

  char* ws = (char*)d_ws;
  float* wts  = (float*)ws;
  int*   cnt  = (int*)(ws + OFF_CNT);
  int*   list = (int*)(ws + OFF_LIST);
  __bf16* W1t = (__bf16*)(ws + OFF_W1T);
  __bf16* W2t = (__bf16*)(ws + OFF_W2T);

  // zero out (includes trailing aux_loss element)
  hipMemsetAsync(d_out, 0, (size_t)out_size * sizeof(float), stream);

  // one-time (per launch) weight convert+transpose to bf16
  transpose_bf16<<<dim3(HDIM / 32, DDIM / 32, NEXP), 256, 0, stream>>>(W1, W1t, DDIM, HDIM);
  transpose_bf16<<<dim3(DDIM / 32, HDIM / 32, NEXP), 256, 0, stream>>>(W2, W2t, HDIM, DDIM);

  moe_router <<<dim3(NTOK * 32 / 256), 256, 0, stream>>>(x, Wr, wts);
  moe_compact<<<dim3(NEXP),            256, 0, stream>>>(wts, cnt, list);
  moe_ffn    <<<dim3(NTOK / 32, NEXP), 256, 0, stream>>>(x, W1t, b1, W2t, b2,
                                                         wts, cnt, list, out);
}
